// SSMBlock_85478439125746
// MI455X (gfx1250) — matrix operand order repaired and checked by reading, not compiled, not run on hardware
//
#include <hip/hip_runtime.h>
#include <hip/hip_bf16.h>

// ---------------------------------------------------------------------------
// SSM block: u = x@B^T ; h = scan(A, u) ; y = h@W^T + b
// bf16 WMMA GEMMs (v_wmma_f32_16x16x32_bf16) with LDS double-buffering fed by
// GLOBAL_LOAD_ASYNC_TO_LDS_B128 (ASYNCcnt), fp32 chunked scan.
// ---------------------------------------------------------------------------

typedef __attribute__((ext_vector_type(16))) __bf16       v16bf;
typedef __attribute__((ext_vector_type(4)))  __bf16       v4bf;
typedef __attribute__((ext_vector_type(8)))  float        v8f;
typedef __attribute__((ext_vector_type(4)))  unsigned int u32x4;

union FragBF {
    v16bf v;
    u32x4 q[2];
};

#define STATE_DIM 1024
#define INPUT_DIM 1024
#define OUTPUT_DIM 1024
#define BATCH 8
#define SEQ 2048
#define NCHUNK 32
#define CHUNK 64   // SEQ / NCHUNK

// GEMM dims (both projections share them)
#define GEMM_M (BATCH * SEQ)   // 16384
#define GEMM_N 1024
#define GEMM_K 1024

// GEMM tiling
#define KSTEP 32
#define NIT   (GEMM_K / KSTEP) // 32
#define TILE_ROWS 128          // block tile M
#define TILE_COLS 128          // block tile N
#define ROW_PITCH_B 80         // bytes per staged row: 32 bf16 (64B) + 16B pad
#define ATILE_BYTES (TILE_ROWS * ROW_PITCH_B)   // 10240
#define BTILE_BYTES (TILE_COLS * ROW_PITCH_B)   // 10240
#define BUF_BYTES   (ATILE_BYTES + BTILE_BYTES) // 20480

// --------------------------- async copy helpers ----------------------------
__device__ __forceinline__ void async_copy_b128(unsigned int lds_off, const void* gaddr) {
    // GLOBAL_LOAD_ASYNC_TO_LDS_B128: vdst = per-lane LDS byte address,
    // vaddr = per-lane 64-bit global address. Tracked with ASYNCcnt.
    asm volatile("global_load_async_to_lds_b128 %0, %1, off"
                 :: "v"(lds_off), "v"(gaddr)
                 : "memory");
}
__device__ __forceinline__ void wait_async_le4() {
    asm volatile("s_wait_asynccnt 0x4" ::: "memory");
}
__device__ __forceinline__ void wait_async_0() {
    asm volatile("s_wait_asynccnt 0x0" ::: "memory");
}

// --------------------------- fp32 -> bf16 convert --------------------------
__global__ __launch_bounds__(256) void cvt_f32_to_bf16(const float* __restrict__ src,
                                                       __bf16* __restrict__ dst, int n) {
    int i = (blockIdx.x * 256 + threadIdx.x) * 4;
    if (i + 3 < n) {
        const float4 f = *reinterpret_cast<const float4*>(src + i);
        v4bf o;
        o.x = (__bf16)f.x; o.y = (__bf16)f.y; o.z = (__bf16)f.z; o.w = (__bf16)f.w;
        *reinterpret_cast<v4bf*>(dst + i) = o;
    }
}

// ----------------------- one K-step of the wave tile -----------------------
// Loads all 6 fragments into distinct registers, then 8 back-to-back WMMAs.
__device__ __forceinline__ void wave_tile_step(const unsigned char* pa0,
                                               const unsigned char* pb0,
                                               v8f acc[4][2]) {
    FragBF bf0, bf1, af0, af1, af2, af3;

    // B fragments: lane l15 = output col in n-tile; lanes 0-15 K[0..8)+[16..24),
    // lanes 16-31 K[8..16)+[24..32).
    bf0.q[0] = *reinterpret_cast<const u32x4*>(pb0);
    bf0.q[1] = *reinterpret_cast<const u32x4*>(pb0 + 32);
    bf1.q[0] = *reinterpret_cast<const u32x4*>(pb0 + 16 * ROW_PITCH_B);
    bf1.q[1] = *reinterpret_cast<const u32x4*>(pb0 + 16 * ROW_PITCH_B + 32);

    // A fragments: lane l15 = row in m-tile; lanes 0-15 K[0..8)+[16..24),
    // lanes 16-31 K[8..16)+[24..32).
    af0.q[0] = *reinterpret_cast<const u32x4*>(pa0);
    af0.q[1] = *reinterpret_cast<const u32x4*>(pa0 + 32);
    af1.q[0] = *reinterpret_cast<const u32x4*>(pa0 + 16 * ROW_PITCH_B);
    af1.q[1] = *reinterpret_cast<const u32x4*>(pa0 + 16 * ROW_PITCH_B + 32);
    af2.q[0] = *reinterpret_cast<const u32x4*>(pa0 + 32 * ROW_PITCH_B);
    af2.q[1] = *reinterpret_cast<const u32x4*>(pa0 + 32 * ROW_PITCH_B + 32);
    af3.q[0] = *reinterpret_cast<const u32x4*>(pa0 + 48 * ROW_PITCH_B);
    af3.q[1] = *reinterpret_cast<const u32x4*>(pa0 + 48 * ROW_PITCH_B + 32);

    acc[0][0] = __builtin_amdgcn_wmma_f32_16x16x32_bf16(false, af0.v, false, bf0.v,
                                                        (short)0, acc[0][0], false, false);
    acc[0][1] = __builtin_amdgcn_wmma_f32_16x16x32_bf16(false, af0.v, false, bf1.v,
                                                        (short)0, acc[0][1], false, false);
    acc[1][0] = __builtin_amdgcn_wmma_f32_16x16x32_bf16(false, af1.v, false, bf0.v,
                                                        (short)0, acc[1][0], false, false);
    acc[1][1] = __builtin_amdgcn_wmma_f32_16x16x32_bf16(false, af1.v, false, bf1.v,
                                                        (short)0, acc[1][1], false, false);
    acc[2][0] = __builtin_amdgcn_wmma_f32_16x16x32_bf16(false, af2.v, false, bf0.v,
                                                        (short)0, acc[2][0], false, false);
    acc[2][1] = __builtin_amdgcn_wmma_f32_16x16x32_bf16(false, af2.v, false, bf1.v,
                                                        (short)0, acc[2][1], false, false);
    acc[3][0] = __builtin_amdgcn_wmma_f32_16x16x32_bf16(false, af3.v, false, bf0.v,
                                                        (short)0, acc[3][0], false, false);
    acc[3][1] = __builtin_amdgcn_wmma_f32_16x16x32_bf16(false, af3.v, false, bf1.v,
                                                        (short)0, acc[3][1], false, false);
}

// ------------------------------- bf16 WMMA GEMM ----------------------------
// C[16384,1024](f32) = Amat[16384,1024](bf16 rm) * Bmat[1024,1024](bf16 rm)^T (+bias)
// 256 threads = 8 waves (2 along M x 4 along N); wave tile 64x32 (4x2 WMMA);
// block tile 128x128. K-slabs staged in LDS via async copies, double buffered
// with a byte-offset XOR switch (keeps one loop body / fixed accumulator regs).
__global__ __launch_bounds__(256) void gemm_bf16_wmma(const __bf16* __restrict__ Amat,
                                                      const __bf16* __restrict__ Bmat,
                                                      float* __restrict__ Cmat,
                                                      const float* __restrict__ bias) {
    __shared__ __align__(16) unsigned char smem[2 * BUF_BYTES];

    const int tid  = threadIdx.x;
    const int lane = tid & 31;
    const int wave = tid >> 5;
    const int wm   = wave & 1;       // 2 waves along M
    const int wn   = wave >> 1;      // 4 waves along N
    const int l15  = lane & 15;
    const int lhi  = lane >> 4;

    const __bf16* Ablk = Amat + (size_t)blockIdx.y * TILE_ROWS * GEMM_K;
    const __bf16* Bblk = Bmat + (size_t)blockIdx.x * TILE_COLS * GEMM_K;

    // Staging: thread t copies 2x16B of A-row (t>>1) and 2x16B of B-row (t>>1);
    // rows are 64B of K-slab, LDS pitch 80B (all-64-bank conflict-free).
    const int srow  = tid >> 1;       // 0..127
    const int shalf = tid & 1;        // 0 or 1 (32B halves)
    const unsigned int smem_base = (unsigned int)(size_t)(&smem[0]);
    const unsigned int ldsA = smem_base + (unsigned int)(srow * ROW_PITCH_B + shalf * 32);
    const unsigned int ldsB = ldsA + ATILE_BYTES;
    const __bf16* gA = Ablk + (size_t)srow * GEMM_K + shalf * 16;
    const __bf16* gB = Bblk + (size_t)srow * GEMM_K + shalf * 16;

    // Per-lane fragment read offsets within a buffer.
    const unsigned int aoff = (unsigned int)((wm * 64 + l15) * ROW_PITCH_B + lhi * 16);
    const unsigned int boff = (unsigned int)(ATILE_BYTES + (wn * 32 + l15) * ROW_PITCH_B + lhi * 16);

    v8f acc[4][2];
#pragma unroll
    for (int i = 0; i < 4; ++i)
#pragma unroll
        for (int j = 0; j < 2; ++j) acc[i][j] = (v8f){};

    // Prologue: stage buffer 0 for kk = 0.
    async_copy_b128(ldsA, gA);
    async_copy_b128(ldsA + 16, gA + 8);
    async_copy_b128(ldsB, gB);
    async_copy_b128(ldsB + 16, gB + 8);

    unsigned int curOff = 0;
    const __bf16* gAn = gA + KSTEP;
    const __bf16* gBn = gB + KSTEP;

    // Steady state: always prefetch next slab, then compute current.
#pragma unroll 1
    for (int it = 0; it < NIT - 1; ++it) {
        const unsigned int nxtOff = curOff ^ BUF_BYTES;
        async_copy_b128(ldsA + nxtOff, gAn);
        async_copy_b128(ldsA + nxtOff + 16, gAn + 8);
        async_copy_b128(ldsB + nxtOff, gBn);
        async_copy_b128(ldsB + nxtOff + 16, gBn + 8);
        gAn += KSTEP;
        gBn += KSTEP;
        wait_async_le4();          // 4 newest = prefetch; current buffer done
        __syncthreads();

        wave_tile_step(smem + curOff + aoff, smem + curOff + boff, acc);
        __syncthreads();           // all reads of current buffer done before re-staging
        curOff = nxtOff;
    }

    // Peeled last iteration (no prefetch).
    wait_async_0();
    __syncthreads();
    wave_tile_step(smem + curOff + aoff, smem + curOff + boff, acc);

    // Epilogue. C/D layout: VGPR r, lane l -> (row base + lhi*8 + r, col base + l15).
    const int mbase = blockIdx.y * TILE_ROWS + wm * 64;
    const int nbase = blockIdx.x * TILE_COLS + wn * 32;
#pragma unroll
    for (int nt = 0; nt < 2; ++nt) {
        const int col = nbase + nt * 16 + l15;
        const float bv = bias ? bias[col] : 0.0f;
#pragma unroll
        for (int mt = 0; mt < 4; ++mt) {
            float* cp = Cmat + (size_t)(mbase + mt * 16 + lhi * 8) * GEMM_N + col;
#pragma unroll
            for (int r = 0; r < 8; ++r)
                cp[(size_t)r * GEMM_N] = acc[mt][nt][r] + bv;
        }
    }
}

// ------------------------------ chunked scan -------------------------------
__global__ __launch_bounds__(256) void scan_local(const float* __restrict__ Avec,
                                                  float* __restrict__ u,
                                                  float* __restrict__ carry) {
    const int g = blockIdx.x * 256 + threadIdx.x;   // [0, BATCH*NCHUNK*STATE_DIM)
    const int d = g & (STATE_DIM - 1);
    const int c = (g >> 10) & (NCHUNK - 1);
    const int b = g >> 15;
    const float a = Avec[d];
    float h = 0.0f;
    size_t base = (((size_t)b * SEQ + (size_t)c * CHUNK) * STATE_DIM) + d;
#pragma unroll 4
    for (int t = 0; t < CHUNK; ++t) {
        const size_t idx = base + (size_t)t * STATE_DIM;
        h = fmaf(a, h, u[idx]);
        u[idx] = h;
    }
    carry[((size_t)b * NCHUNK + c) * STATE_DIM + d] = h;
}

__global__ __launch_bounds__(256) void scan_carry(const float* __restrict__ Avec,
                                                  float* __restrict__ carry) {
    const int g = blockIdx.x * 256 + threadIdx.x;   // [0, BATCH*STATE_DIM)
    const int d = g & (STATE_DIM - 1);
    const int b = g >> 10;
    float a64 = Avec[d];
#pragma unroll
    for (int i = 0; i < 6; ++i) a64 *= a64;         // A^64
    float C = 0.0f;
    for (int c = 0; c < NCHUNK; ++c) {
        const size_t idx = ((size_t)b * NCHUNK + c) * STATE_DIM + d;
        C = fmaf(a64, C, carry[idx]);
        carry[idx] = C;                             // inclusive prefix
    }
}

__global__ __launch_bounds__(256) void scan_apply(const float* __restrict__ Avec,
                                                  const float* __restrict__ u,
                                                  const float* __restrict__ carry,
                                                  __bf16* __restrict__ hb) {
    const int g = blockIdx.x * 256 + threadIdx.x;
    const int d = g & (STATE_DIM - 1);
    const int c = (g >> 10) & (NCHUNK - 1);
    const int b = g >> 15;
    const float a = Avec[d];
    size_t base = (((size_t)b * SEQ + (size_t)c * CHUNK) * STATE_DIM) + d;
    if (c == 0) {
#pragma unroll 4
        for (int t = 0; t < CHUNK; ++t) {
            const size_t idx = base + (size_t)t * STATE_DIM;
            hb[idx] = (__bf16)u[idx];
        }
    } else {
        const float Cp = carry[((size_t)b * NCHUNK + (c - 1)) * STATE_DIM + d];
        float p = a;
#pragma unroll 4
        for (int t = 0; t < CHUNK; ++t) {
            const size_t idx = base + (size_t)t * STATE_DIM;
            hb[idx] = (__bf16)fmaf(p, Cp, u[idx]);
            p *= a;
        }
    }
}

// ------------------------------- launcher ----------------------------------
extern "C" void kernel_launch(void* const* d_in, const int* in_sizes, int n_in,
                              void* d_out, int out_size, void* d_ws, size_t ws_size,
                              hipStream_t stream) {
    const float* x    = (const float*)d_in[0];  // [8,2048,1024]
    const float* Avec = (const float*)d_in[1];  // [1024]
    const float* Bmat = (const float*)d_in[2];  // [1024,1024]
    const float* Wmat = (const float*)d_in[3];  // [1024,1024]
    const float* bvec = (const float*)d_in[4];  // [1024]
    float* out = (float*)d_out;                 // [8,2048,1024]

    const int NX = BATCH * SEQ * INPUT_DIM;     // 16,777,216
    const int NW = OUTPUT_DIM * STATE_DIM;      // 1,048,576

    char* ws = (char*)d_ws;
    size_t off = 0;
    __bf16* xb = (__bf16*)(ws + off); off += (size_t)NX * 2;            // 32 MiB
    __bf16* Bb = (__bf16*)(ws + off); off += (size_t)NW * 2;            //  2 MiB
    __bf16* Wb = (__bf16*)(ws + off); off += (size_t)NW * 2;            //  2 MiB
    float*  u  = (float*)(ws + off);  off += (size_t)NX * 4;            // 64 MiB
    __bf16* hb = (__bf16*)(ws + off); off += (size_t)NX * 2;            // 32 MiB
    float*  cr = (float*)(ws + off);  off += (size_t)BATCH * NCHUNK * STATE_DIM * 4; // 1 MiB
    (void)ws_size; (void)in_sizes; (void)n_in; (void)out_size;

    // 1) convert x, B, W to bf16
    cvt_f32_to_bf16<<<NX / 1024, 256, 0, stream>>>(x, xb, NX);
    cvt_f32_to_bf16<<<NW / 1024, 256, 0, stream>>>(Bmat, Bb, NW);
    cvt_f32_to_bf16<<<NW / 1024, 256, 0, stream>>>(Wmat, Wb, NW);

    // 2) u = x @ B^T   (f32 accumulate/output)
    dim3 g1(GEMM_N / TILE_COLS, GEMM_M / TILE_ROWS);
    gemm_bf16_wmma<<<g1, 256, 0, stream>>>(xb, Bb, u, nullptr);

    // 3) chunked diagonal scan: h = scan(A, u); emit bf16 h
    scan_local<<<(BATCH * NCHUNK * STATE_DIM) / 256, 256, 0, stream>>>(Avec, u, cr);
    scan_carry<<<(BATCH * STATE_DIM) / 256, 256, 0, stream>>>(Avec, cr);
    scan_apply<<<(BATCH * NCHUNK * STATE_DIM) / 256, 256, 0, stream>>>(Avec, u, cr, hb);

    // 4) y = h @ W^T + b
    dim3 g2(GEMM_N / TILE_COLS, GEMM_M / TILE_ROWS);
    gemm_bf16_wmma<<<g2, 256, 0, stream>>>(hb, Wb, out, bvec);
}
